// QuantizedLinearReLU_47313359733170
// MI455X (gfx1250) — 8-bit matrix operand order repaired and checked by reading, compiled, not run on hardware
//
#include <hip/hip_runtime.h>

typedef int v8i __attribute__((ext_vector_type(8)));

#define IN_F  256
#define OUT_F 256
#define ROWS_PER_BLOCK 128
#define A_STRIDE 67
#define A_DWORDS (ROWS_PER_BLOCK * A_STRIDE)

__global__ __launch_bounds__(256) void pack_weights(const int* __restrict__ W,
                                                    int* __restrict__ bswz) {
    int idx   = blockIdx.x * 256 + threadIdx.x;
    int v     = idx & 7;
    int lane  = (idx >> 3) & 31;
    int chunk = idx >> 8;
    int kc    = chunk & 3;
    int nt    = chunk >> 2;
    int n = nt * 16 + (lane & 15);
    int k = kc * 64 + (v & 3) * 4 + (v >> 2) * 32 + ((lane >> 4) << 4);
    const int* wr = W + n * IN_F + k;
    bswz[idx] = (wr[0] & 255) | ((wr[1] & 255) << 8) |
                ((wr[2] & 255) << 16) | ((wr[3] & 255) << 24);
}

__device__ __forceinline__ int quant_pack4(float4 f) {
    int q0 = (int)rintf(f.x * 20.0f); q0 = q0 < -128 ? -128 : (q0 > 127 ? 127 : q0);
    int q1 = (int)rintf(f.y * 20.0f); q1 = q1 < -128 ? -128 : (q1 > 127 ? 127 : q1);
    int q2 = (int)rintf(f.z * 20.0f); q2 = q2 < -128 ? -128 : (q2 > 127 ? 127 : q2);
    int q3 = (int)rintf(f.w * 20.0f); q3 = q3 < -128 ? -128 : (q3 > 127 ? 127 : q3);
    return (q0 & 255) | ((q1 & 255) << 8) | ((q2 & 255) << 16) | ((q3 & 255) << 24);
}

__device__ __forceinline__ int requant_shift31(int sf, int acc) {
    int      hi = __mulhi(sf, acc);
    unsigned lo = (unsigned)(sf * acc);
    return (int)((lo >> 31) | ((unsigned)hi << 1));
}

__device__ __forceinline__ float requant_out(int sf, int acc) {
    int p = requant_shift31(sf, acc);
    int q = p < 0 ? 0 : (p > 255 ? 255 : p);
    return (float)q * 0.1f;
}

__global__ __launch_bounds__(256) void qlinear_relu(
        const float* __restrict__ x,
        const int*   __restrict__ bswz,
        const int*   __restrict__ scale_factor,
        const int*   __restrict__ q_bias,
        float*       __restrict__ out) {
    __shared__ int aq[A_DWORDS];

    const int tid = threadIdx.x;
    const int m0  = blockIdx.x * ROWS_PER_BLOCK;

    const float4* xv = (const float4*)x + (size_t)m0 * (IN_F / 4);
    #pragma unroll
    for (int i = 0; i < 32; ++i) {
        int nidx = i * 256 + tid;
        int m    = nidx >> 6;
        int k4   = nidx & 63;
        aq[m * A_STRIDE + k4] = quant_pack4(xv[m * (IN_F / 4) + k4]);
    }
    __syncthreads();

    const int wave = tid >> 5;
    const int lane = tid & 31;
    const int lh   = lane & 15;
    const int hi   = lane >> 4;

    v8i A[4];
    {
        const int rbase = (wave * 16 + lh) * A_STRIDE;
        #pragma unroll
        for (int kc = 0; kc < 4; ++kc) {
            #pragma unroll
            for (int v = 0; v < 8; ++v) {
                int k = kc * 64 + (v & 3) * 4 + (v >> 2) * 32 + hi * 16;
                A[kc][v] = aq[rbase + (k >> 2)];
            }
        }
    }

    const int4* bp4   = (const int4*)bswz;
    float*      obase = out + (size_t)(m0 + wave * 16) * OUT_F;

    for (int nt = 0; nt < 16; nt += 2) {
        const int n0  = nt * 16 + lh;
        const int n1  = n0 + 16;
        const int qb0 = q_bias[n0];
        const int qb1 = q_bias[n1];
        const int sf0 = scale_factor[n0];
        const int sf1 = scale_factor[n1];

        v8i acc0 = { qb0, qb0, qb0, qb0, qb0, qb0, qb0, qb0 };
        v8i acc1 = { qb1, qb1, qb1, qb1, qb1, qb1, qb1, qb1 };

        #pragma unroll
        for (int kc = 0; kc < 4; ++kc) {
            const int4 a0 = bp4[(nt * 4 + kc) * 64 + lane * 2];
            const int4 a1 = bp4[(nt * 4 + kc) * 64 + lane * 2 + 1];
            const int4 b0 = bp4[((nt + 1) * 4 + kc) * 64 + lane * 2];
            const int4 b1 = bp4[((nt + 1) * 4 + kc) * 64 + lane * 2 + 1];
            v8i B0 = { a0.x, a0.y, a0.z, a0.w, a1.x, a1.y, a1.z, a1.w };
            v8i B1 = { b0.x, b0.y, b0.z, b0.w, b1.x, b1.y, b1.z, b1.w };
            acc0 = __builtin_amdgcn_wmma_i32_16x16x64_iu8(true, A[kc], true, B0, acc0,
                                                          false, false);
            acc1 = __builtin_amdgcn_wmma_i32_16x16x64_iu8(true, A[kc], true, B1, acc1,
                                                          false, false);
        }

        #pragma unroll
        for (int j = 0; j < 8; ++j) {
            float r0 = requant_out(sf0, acc0[j]);
            float r1 = requant_out(sf1, acc1[j]);
            float* orow = obase + (size_t)(j + hi * 8) * OUT_F;
            orow[n0] = r0;
            orow[n1] = r1;
        }
    }
}

extern "C" void kernel_launch(void* const* d_in, const int* in_sizes, int n_in,
                              void* d_out, int out_size, void* d_ws, size_t ws_size,
                              hipStream_t stream) {
    const float* x  = (const float*)d_in[0];
    const int*   W  = (const int*)d_in[1];
    const int*   sf = (const int*)d_in[2];
    const int*   qb = (const int*)d_in[3];
    float*       out = (float*)d_out;
    int*         bswz = (int*)d_ws;

    pack_weights<<<64, 256, 0, stream>>>(W, bswz);

    const int rows = in_sizes[0] / IN_F;
    qlinear_relu<<<rows / ROWS_PER_BLOCK, 256, 0, stream>>>(x, bswz, sf, qb, out);
}
